// MoveEncoder_24154896073283
// MI455X (gfx1250) — compile-verified, matrix operand order checked by reading, not run on hardware
//
#include <hip/hip_runtime.h>
#include <hip/hip_bf16.h>

// MoveEncoder fused kernel for MI455X (gfx1250, wave32, WMMA, async-LDS DMA).
//   out[m,e] = 2*( (sum_k cs_k x[m,k]) @ Wv^T + 15*bv )[e],
//   cs_k = column sums of softmax(q k^T); V is never materialized.
// Prep kernel: f16-convert Wq/Wk/Wv and precompute XT[l][tok] = f16(letter+pos)
// (512 rows only) so the main kernel's staging is a pure f16 copy -> async DMA.

typedef __attribute__((ext_vector_type(16))) _Float16 v16h;
typedef __attribute__((ext_vector_type(8)))  _Float16 h8;
typedef __attribute__((ext_vector_type(8)))  float    v8f;
typedef int v4i_ __attribute__((vector_size(16)));     // b128 pointee for async DMA

#define M_MOVES 4096
#define LW      15
#define DIM     1024
#define MV_PER_WG 8          // 8 moves * 16 padded rows = 128 rows / workgroup

// f16 workspace layout (element offsets)
#define WQH_OFF 0
#define WKH_OFF (1024*1024)
#define WVH_OFF (2*1024*1024)
#define XT_OFF  (3*1024*1024)          // [16][32][1024], row l=15 zeroed
#define WS_F16_ELEMS (3*1024*1024 + 512*1024)

// LDS layout (bytes). Padded strides for bank-conflict-free ds_load_b128.
#define TILE_LD   40                   // [128][40] halves (80B row pitch)
#define TILE_BYTES (128*TILE_LD*2)     // 10240
#define XS_OFF   0                     // 2 buffers
#define WQS_OFF  (2*TILE_BYTES)        // 20480, 2 buffers
#define WKS_OFF  (4*TILE_BYTES)        // 40960, 2 buffers
#define QKS_OFF  (6*TILE_BYTES)        // 61440: 8 waves x (Q[16][136],K[16][136])
#define QK_LD    136
#define QK_WAVE_ELEMS (2*16*QK_LD)     // 4352 halves = 8704 B per wave
#define CS_OFF   (QKS_OFF + 8*QK_WAVE_ELEMS*2)   // 131072
#define YS_LD    1032                  // phase B: Ys[16][1032] at byte 0
#define WVS_OFF  49152                 // phase B: Wvs[1024][40] (81920 B)
#define LDS_BYTES (CS_OFF + 512)

#if defined(__HIP_DEVICE_COMPILE__) && __has_builtin(__builtin_amdgcn_global_load_async_to_lds_b128)
#define HAVE_ASYNC 1
#else
#define HAVE_ASYNC 0
#endif

#if HAVE_ASYNC
#if __has_builtin(__builtin_amdgcn_s_wait_asynccnt)
#define WAIT_ASYNC(N) __builtin_amdgcn_s_wait_asynccnt(N)
#else
#define WAIT_ASYNC(N) asm volatile("s_wait_asynccnt " #N ::: "memory")
#endif
#else
#define WAIT_ASYNC(N) ((void)0)
#endif

__device__ __forceinline__ void async_copy16(const _Float16* gsrc, _Float16* ldst) {
#if HAVE_ASYNC
  __builtin_amdgcn_global_load_async_to_lds_b128(
      (__attribute__((address_space(1))) v4i_*)gsrc,
      (__attribute__((address_space(3))) v4i_*)ldst, 0, 0);
#else
  *(h8*)ldst = *(const h8*)gsrc;     // synchronous fallback (same barriers)
#endif
}

__device__ __forceinline__ h8 cvt8(const float* __restrict__ a) {
  h8 r;
#pragma unroll
  for (int i = 0; i < 8; ++i) r[i] = (_Float16)a[i];
  return r;
}
__device__ __forceinline__ h8 add_cvt8(const float* __restrict__ a,
                                       const float* __restrict__ b) {
  h8 r;
#pragma unroll
  for (int i = 0; i < 8; ++i) r[i] = (_Float16)(a[i] + b[i]);
  return r;
}

// A/B fragment loader for v_wmma_f32_16x16x32_f16: lane t covers row/col (t%16),
// 8 halves at K=kbase+(t/16)*8 and 8 more at K=kbase+16+(t/16)*8 (ISA 7.12.2).
__device__ __forceinline__ v16h load_frag(const _Float16* buf, int ld,
                                          int rowbase, int kbase, int lane) {
  int r  = rowbase + (lane & 15);
  int k0 = kbase + ((lane >> 4) << 3);
  h8 lo = *(const h8*)(buf + r * ld + k0);
  h8 hi = *(const h8*)(buf + r * ld + k0 + 16);
  v16h out;
#pragma unroll
  for (int i = 0; i < 8; ++i) { out[i] = lo[i]; out[i + 8] = hi[i]; }
  return out;
}

// ---------------- prep: f32 -> f16 weights + fused embedding table ----------
__global__ __launch_bounds__(256)
void prep_kernel(const float* __restrict__ letter_emb,
                 const float* __restrict__ pos_emb,
                 const float* __restrict__ Wq,
                 const float* __restrict__ Wk,
                 const float* __restrict__ Wv,
                 _Float16*    __restrict__ ws)
{
  const int WCH = (1024 * 1024) / 8;                 // chunks per weight
  int c = blockIdx.x * 256 + threadIdx.x;            // one h8 chunk each
  if (c < 3 * WCH) {
    const float* src = (c < WCH) ? Wq : (c < 2 * WCH ? Wk : Wv);
    int cc = (c < WCH) ? c : (c < 2 * WCH ? c - WCH : c - 2 * WCH);
    *(h8*)(ws + (size_t)(c / WCH) * (1024 * 1024) + cc * 8) = cvt8(src + cc * 8);
  } else if (c < 3 * WCH + 512 * 128) {
    int c2 = c - 3 * WCH;                            // XT: 512 rows x 128 chunks
    int d8 = c2 & 127, row = c2 >> 7;                // row = l*32 + tok
    int l = row >> 5, tok = row & 31;
    h8 v = {};
    if (l < LW)
      v = add_cvt8(letter_emb + tok * DIM + d8 * 8, pos_emb + l * DIM + d8 * 8);
    *(h8*)(ws + XT_OFF + (size_t)row * DIM + d8 * 8) = v;
  }
}

// ---------------- main fused kernel ----------------------------------------
__global__ __launch_bounds__(256)
void move_encoder_kernel(const int*      __restrict__ tokens,
                         const _Float16* __restrict__ XT,
                         const _Float16* __restrict__ WqH,
                         const _Float16* __restrict__ WkH,
                         const _Float16* __restrict__ WvH,
                         const float*    __restrict__ bq,
                         const float*    __restrict__ bk,
                         const float*    __restrict__ bv,
                         float*          __restrict__ out)
{
  extern __shared__ unsigned char smem[];

  const int tid  = threadIdx.x;
  const int lane = tid & 31;
  const int wid  = tid >> 5;
  const int g    = lane >> 4;
  const int n    = lane & 15;
  const int m0   = blockIdx.x * MV_PER_WG;

  _Float16* Xs  = (_Float16*)(smem + XS_OFF);
  _Float16* Wqs = (_Float16*)(smem + WQS_OFF);
  _Float16* Wks = (_Float16*)(smem + WKS_OFF);
  _Float16* Qh  = (_Float16*)(smem + QKS_OFF) + wid * QK_WAVE_ELEMS;
  _Float16* Kh  = Qh + 16 * QK_LD;
  float*    csbuf = (float*)(smem + CS_OFF);

  // Per-thread fixed staging assignment: one 128-row tile row, two 16B segs.
  const int arow = tid >> 1;               // 0..127
  const int aq0  = (tid & 1) * 2;          // segments aq0, aq0+1 (of 4)
  const int al = arow & 15, amv = arow >> 4;
  int atok = 0;
  if (al < LW) atok = tokens[(m0 + amv) * LW + al];
  const _Float16* xrow = XT + (size_t)(al * 32 + atok) * DIM;   // l==15 -> zero row

  auto issue_a = [&](int i) {              // stage tile for flat iter i
    int et = i >> 5, kt = i & 31, buf = i & 1;
    _Float16* XsB = Xs  + buf * (128 * TILE_LD);
    _Float16* WqB = Wqs + buf * (128 * TILE_LD);
    _Float16* WkB = Wks + buf * (128 * TILE_LD);
    int e = et * 128 + arow;
    const _Float16* xs = xrow + kt * 32;
    const _Float16* qs = WqH + (size_t)e * DIM + kt * 32;
    const _Float16* ks = WkH + (size_t)e * DIM + kt * 32;
#pragma unroll
    for (int j = 0; j < 2; ++j) {
      int q = aq0 + j;
      async_copy16(xs + q * 8, XsB + arow * TILE_LD + q * 8);
      async_copy16(qs + q * 8, WqB + arow * TILE_LD + q * 8);
      async_copy16(ks + q * 8, WkB + arow * TILE_LD + q * 8);
    }
  };

  v8f S = {};                 // 16x16 score accumulator (this wave's move)
  v8f Qacc[8], Kacc[8];

  issue_a(0);
  for (int i = 0; i < 256; ++i) {          // et = i>>5 (8 e-tiles), kt = i&31
    int et = i >> 5, kt = i & 31, buf = i & 1;
    if (kt == 0) {
      v8f z = {};
#pragma unroll
      for (int f = 0; f < 8; ++f) { Qacc[f] = z; Kacc[f] = z; }
    }
    __syncthreads();                       // all reads of buf^1 done
    if (i + 1 < 256) { issue_a(i + 1); WAIT_ASYNC(6); }
    else             { WAIT_ASYNC(0); }
    __syncthreads();                       // tile i landed for all waves

    const _Float16* XsB = Xs  + buf * (128 * TILE_LD);
    const _Float16* WqB = Wqs + buf * (128 * TILE_LD);
    const _Float16* WkB = Wks + buf * (128 * TILE_LD);

    // Software-pipelined fragment fetch: issue nf+1's B loads before consuming
    // nf's, so the waitcnt pass can use partial s_wait_dscnt instead of 0.
    v16h A   = load_frag(XsB, TILE_LD, wid * 16, 0, lane);
    v16h Bq0 = load_frag(WqB, TILE_LD, 0, 0, lane);
    v16h Bk0 = load_frag(WkB, TILE_LD, 0, 0, lane);
#pragma unroll
    for (int nf = 0; nf < 8; ++nf) {
      v16h BqN = Bq0, BkN = Bk0;
      if (nf < 7) {
        BqN = load_frag(WqB, TILE_LD, (nf + 1) * 16, 0, lane);
        BkN = load_frag(WkB, TILE_LD, (nf + 1) * 16, 0, lane);
      }
      Qacc[nf] = __builtin_amdgcn_wmma_f32_16x16x32_f16(
          false, A, false, Bq0, (short)0, Qacc[nf], false, false);
      Kacc[nf] = __builtin_amdgcn_wmma_f32_16x16x32_f16(
          false, A, false, Bk0, (short)0, Kacc[nf], false, false);
      Bq0 = BqN; Bk0 = BkN;
    }

    if (kt == 31) {
      // bias + stage q,k as f16 in wave-private LDS (C-layout map), then
      // S += Q K^T over this 128-wide e-tile.
#pragma unroll
      for (int nf = 0; nf < 8; ++nf) {
        int e = et * 128 + nf * 16 + n;
        float qb = bq[e], kb = bk[e];
#pragma unroll
        for (int r = 0; r < 8; ++r) {
          int row = r + 8 * g;
          Qh[row * QK_LD + nf * 16 + n] = (_Float16)(Qacc[nf][r] + qb);
          Kh[row * QK_LD + nf * 16 + n] = (_Float16)(Kacc[nf][r] + kb);
        }
      }
#pragma unroll
      for (int c = 0; c < 4; ++c) {
        v16h Aq  = load_frag(Qh, QK_LD, 0, c * 32, lane);
        v16h Bk2 = load_frag(Kh, QK_LD, 0, c * 32, lane);
        S = __builtin_amdgcn_wmma_f32_16x16x32_f16(
            false, Aq, false, Bk2, (short)0, S, false, false);
      }
    }
  }

  // Softmax over cols (mask padded col 15), column sums over rows 0..14.
  float p = 0.0f;
#pragma unroll
  for (int r = 0; r < 8; ++r) {
    float v = (n == 15) ? -3.0e38f : S[r];
    float mx = v;
#pragma unroll
    for (int off = 1; off < 16; off <<= 1) mx = fmaxf(mx, __shfl_xor(mx, off, 32));
    float e = __expf(v - mx);
    float sum = e;
#pragma unroll
    for (int off = 1; off < 16; off <<= 1) sum += __shfl_xor(sum, off, 32);
    if ((r + 8 * g) != 15) p += e / sum;
  }
  p += __shfl_xor(p, 16, 32);
  if (lane < 16) csbuf[wid * 16 + n] = p;

  __syncthreads();                         // phase-A LDS dead; reuse buffers

  // ---- Phase B: out = 2*(Y @ Wv^T) + 30*bv,  Y row = sum_k cs_k * XT row ----
  _Float16* Ys  = (_Float16*)(smem);                 // [16][YS_LD]
  _Float16* Wvs = (_Float16*)(smem + WVS_OFF);       // [1024][40]
  {
    int dbase = lane * 32;
    float acc[32];
#pragma unroll
    for (int j = 0; j < 32; ++j) acc[j] = 0.0f;
    for (int k = 0; k < LW; ++k) {
      float cs = csbuf[wid * 16 + k];
      int tok = tokens[(m0 + wid) * LW + k];
      const _Float16* xr = XT + (size_t)(k * 32 + tok) * DIM + dbase;
#pragma unroll
      for (int c = 0; c < 4; ++c) {
        h8 hv = *(const h8*)(xr + c * 8);
#pragma unroll
        for (int j = 0; j < 8; ++j) acc[c * 8 + j] += cs * (float)hv[j];
      }
    }
#pragma unroll
    for (int c = 0; c < 4; ++c) {
      h8 hv;
#pragma unroll
      for (int j = 0; j < 8; ++j) hv[j] = (_Float16)acc[c * 8 + j];
      *(h8*)(Ys + wid * YS_LD + dbase + c * 8) = hv;
    }
    h8 z = {};
    int zrow = 8 + wid;                              // zero pad rows 8..15
#pragma unroll
    for (int c = 0; c < 4; ++c)
      *(h8*)(Ys + zrow * YS_LD + dbase + c * 8) = z;
  }

  v8f Cv[8] = {};
  for (int kt = 0; kt < 32; ++kt) {
    __syncthreads();
#pragma unroll
    for (int it = 0; it < 16; ++it) {                // 4096 chunks / 256 thr
      int idx = tid + it * 256;
      int row = idx >> 2, q = idx & 3;
      async_copy16(WvH + (size_t)row * DIM + kt * 32 + q * 8,
                   Wvs + row * TILE_LD + q * 8);
    }
    WAIT_ASYNC(0);
    __syncthreads();
    v16h Av = load_frag(Ys, YS_LD, 0, kt * 32, lane);
#pragma unroll
    for (int nf = 0; nf < 8; ++nf) {
      v16h Bv = load_frag(Wvs, TILE_LD, wid * 128 + nf * 16, 0, lane);
      Cv[nf] = __builtin_amdgcn_wmma_f32_16x16x32_f16(
          false, Av, false, Bv, (short)0, Cv[nf], false, false);
    }
  }

  if (lane < 16) {                                   // rows 0..7 = moves
#pragma unroll
    for (int nf = 0; nf < 8; ++nf) {
      int e = wid * 128 + nf * 16 + n;
      float bterm = 30.0f * bv[e];                   // 2 * 15 * bv
#pragma unroll
      for (int r = 0; r < 8; ++r)
        out[(size_t)(m0 + r) * DIM + e] = 2.0f * Cv[nf][r] + bterm;
    }
  }
}

extern "C" void kernel_launch(void* const* d_in, const int* in_sizes, int n_in,
                              void* d_out, int out_size, void* d_ws, size_t ws_size,
                              hipStream_t stream) {
  const int*   tokens     = (const int*)d_in[0];
  const float* letter_emb = (const float*)d_in[1];
  const float* pos_emb    = (const float*)d_in[2];
  const float* Wq         = (const float*)d_in[3];
  const float* bq         = (const float*)d_in[4];
  const float* Wk         = (const float*)d_in[5];
  const float* bk         = (const float*)d_in[6];
  const float* Wv         = (const float*)d_in[7];
  const float* bv         = (const float*)d_in[8];
  float* out = (float*)d_out;
  _Float16* ws = (_Float16*)d_ws;        // needs 7 MB (WS_F16_ELEMS * 2 bytes)

  // prep: 3*131072 weight chunks + 65536 XT chunks = 458752 -> 1792 blocks
  prep_kernel<<<dim3(1792), 256, 0, stream>>>(letter_emb, pos_emb, Wq, Wk, Wv, ws);

  move_encoder_kernel<<<dim3(M_MOVES / MV_PER_WG), 256, LDS_BYTES, stream>>>(
      tokens, ws + XT_OFF, ws + WQH_OFF, ws + WKH_OFF, ws + WVH_OFF,
      bq, bk, bv, out);
}
